// HybridSerialClassifier_50208167690601
// MI455X (gfx1250) — matrix operand order repaired and checked by reading, not compiled, not run on hardware
//
#include <hip/hip_runtime.h>
#include <hip/hip_bf16.h>

typedef __attribute__((ext_vector_type(16))) _Float16 v16h;
typedef __attribute__((ext_vector_type(8)))  _Float16 v8h;
typedef __attribute__((ext_vector_type(8)))  float    v8f;

#define D_STATE   16
#define D_INNER   256
#define DT_RANK   8
#define LSTM_H    128
#define NUM_CLASSES 5
#define IN_CH     12
#define Bsz       32
#define Lseq      1024
#define NTOK      (Bsz*Lseq)
#define XDBC_LD   48   // 40 padded to 48 (3 WMMA n-tiles)

// ---------------- fast, branch-free nonlinearities (v_exp_f32 + v_rcp_f32) ----------------
__device__ __forceinline__ float fsigm(float x) {
  return __builtin_amdgcn_rcpf(1.f + __expf(-x));
}
__device__ __forceinline__ float ftanh(float x) {
  // tanh(x) = 1 - 2/(exp(2x)+1); exp->inf gives rcp->0 -> +1, exp->0 gives -1
  return 1.f - 2.f * __builtin_amdgcn_rcpf(__expf(2.f * x) + 1.f);
}
__device__ __forceinline__ float fsilu(float x) { return x * fsigm(x); }

__device__ __forceinline__ v8f zero8() {
  v8f z;
#pragma unroll
  for (int i = 0; i < 8; ++i) z[i] = 0.f;
  return z;
}

// A fragment (16x32 f16, MxK): lane (0..15)=row m, half=lane>>4.
// elems 0..7 -> K = half*8 + e ; elems 8..15 -> K = 16 + half*8 + e
__device__ __forceinline__ v16h load_a(const _Float16* p0, int ld) {
  int lane = threadIdx.x & 31;
  const _Float16* p = p0 + (size_t)(lane & 15) * ld + (lane >> 4) * 8;
  v8h lo = *(const v8h*)(p);
  v8h hi = *(const v8h*)(p + 16);
  v16h r;
#pragma unroll
  for (int i = 0; i < 8; ++i) { r[i] = lo[i]; r[i + 8] = hi[i]; }
  return r;
}

// B fragment (32x16 f16, KxN) from row-major weight W[N][K]:
// lane = column n, elems 0..7 -> K = half*8 + e ; elems 8..15 -> K = 16 + half*8 + e
__device__ __forceinline__ v16h load_b(const _Float16* p0, int ld) {
  int lane = threadIdx.x & 31;
  const _Float16* p = p0 + (size_t)(lane & 15) * ld + (lane >> 4) * 8;
  v8h lo = *(const v8h*)(p);
  v8h hi = *(const v8h*)(p + 16);
  v16h r;
#pragma unroll
  for (int i = 0; i < 8; ++i) { r[i] = lo[i]; r[i + 8] = hi[i]; }
  return r;
}

#define WMMA_F16(a, b, c) \
  __builtin_amdgcn_wmma_f32_16x16x32_f16(false, (a), false, (b), (short)0, (c), false, false)

// ---------------- prep kernels ----------------
__global__ void k_f32_to_f16(const float* __restrict__ s, _Float16* __restrict__ d, int n) {
  int i = blockIdx.x * blockDim.x + threadIdx.x;
  if (i < n) d[i] = (_Float16)s[i];
}

__global__ void k_pad_xproj(const float* __restrict__ s, _Float16* __restrict__ d) {
  int i = blockIdx.x * blockDim.x + threadIdx.x;           // 48*256
  if (i >= XDBC_LD * D_INNER) return;
  int row = i / D_INNER, col = i % D_INNER;
  d[i] = (row < DT_RANK + 2 * D_STATE) ? (_Float16)s[row * D_INNER + col] : (_Float16)0.f;
}

__global__ void k_bias_sum(const float* __restrict__ a, const float* __restrict__ b,
                           float* __restrict__ o, int n) {
  int i = blockIdx.x * blockDim.x + threadIdx.x;
  if (i < n) o[i] = a[i] + b[i];
}

// ---------------- h = x @ w_proj^T + b_proj  (K=12, scalar) ----------------
__global__ void k_proj(const float* __restrict__ x, const float* __restrict__ w,
                       const float* __restrict__ b, _Float16* __restrict__ h16) {
  int i = blockIdx.x * blockDim.x + threadIdx.x;           // NTOK*128
  if (i >= NTOK * 128) return;
  int d = i & 127, tok = i >> 7;
  float acc = b[d];
  const float* xr = x + (size_t)tok * IN_CH;
  const float* wr = w + (size_t)d * IN_CH;
#pragma unroll
  for (int c = 0; c < IN_CH; ++c) acc += xr[c] * wr[c];
  h16[i] = (_Float16)acc;
}

// ---------------- generic WMMA GEMM: C = A(MxK,f16) @ W(NxK,f16)^T + bias ----------------
// block = 128 threads (4 waves); wave w -> m-tile blockIdx.x*4+w; NT n-tiles per wave.
template <int NT>
__global__ __launch_bounds__(128) void k_gemm(const _Float16* __restrict__ A, int lda,
                                              const _Float16* __restrict__ W, int ldb,
                                              const float* __restrict__ bias,
                                              float* __restrict__ Cf, _Float16* __restrict__ Ch,
                                              int ldc, int K) {
  int wave = threadIdx.x >> 5;
  int lane = threadIdx.x & 31;
  int mt = blockIdx.x * 4 + wave;
  int n0 = blockIdx.y * (16 * NT);
  const _Float16* Abase = A + (size_t)mt * 16 * lda;
  v8f acc[NT];
#pragma unroll
  for (int j = 0; j < NT; ++j) acc[j] = zero8();
  for (int k0 = 0; k0 < K; k0 += 32) {
    if (k0 + 32 < K) __builtin_prefetch(Abase + k0 + 32, 0, 0);  // global_prefetch
    v16h a = load_a(Abase + k0, lda);
#pragma unroll
    for (int j = 0; j < NT; ++j) {
      v16h bf = load_b(W + (size_t)(n0 + j * 16) * ldb + k0, ldb);
      acc[j] = WMMA_F16(a, bf, acc[j]);
    }
  }
  int row0 = mt * 16 + (lane >> 4) * 8;
  int colb = lane & 15;
#pragma unroll
  for (int j = 0; j < NT; ++j) {
    int c = n0 + j * 16 + colb;
    float bv = bias ? bias[c] : 0.f;
#pragma unroll
    for (int r = 0; r < 8; ++r) {
      float v = acc[j][r] + bv;
      size_t idx = (size_t)(row0 + r) * ldc + c;
      if (Cf) Cf[idx] = v;
      if (Ch) Ch[idx] = (_Float16)v;
    }
  }
}

// ---------------- causal depthwise conv (D_CONV=4) + SiLU ----------------
__global__ void k_conv_silu(const float* __restrict__ xz, const float* __restrict__ cw,
                            const float* __restrict__ cb, float* __restrict__ u32o,
                            _Float16* __restrict__ u16o) {
  int i = blockIdx.x * blockDim.x + threadIdx.x;           // NTOK*256
  if (i >= NTOK * D_INNER) return;
  int d = i & 255, tok = i >> 8, l = tok & (Lseq - 1);
  float acc = cb[d];
#pragma unroll
  for (int k = 0; k < 4; ++k) {
    int lo = l + k - 3;
    if (lo >= 0) acc += cw[d * 4 + k] * xz[(size_t)(tok + k - 3) * 512 + d];
  }
  float s = fsilu(acc);
  u32o[i] = s;
  u16o[i] = (_Float16)s;
}

// ---------------- dt = softplus(xdbc[:, :8] @ w_dt^T + b_dt)  (K=8, scalar) ----------------
__global__ void k_dt(const float* __restrict__ xdbc, const float* __restrict__ wdt,
                     const float* __restrict__ bdt, float* __restrict__ dt) {
  int i = blockIdx.x * blockDim.x + threadIdx.x;           // NTOK*256
  if (i >= NTOK * D_INNER) return;
  int d = i & 255, tok = i >> 8;
  float acc = bdt[d];
  const float* xr = xdbc + (size_t)tok * XDBC_LD;
#pragma unroll
  for (int r = 0; r < DT_RANK; ++r) acc += xr[r] * wdt[d * DT_RANK + r];
  dt[i] = (acc > 20.f) ? acc : __logf(1.f + __expf(acc));
}

// ---------------- selective-scan (Mamba SSM), one block per batch ----------------
// Software-pipelined: B/C double-buffered in LDS, next-step dt/u/z prefetched into
// registers, single barrier per step.
__global__ __launch_bounds__(256) void k_scan(const float* __restrict__ xdbc,
                                              const float* __restrict__ dt,
                                              const float* __restrict__ u32,
                                              const float* __restrict__ xz,
                                              const float* __restrict__ A_log,
                                              const float* __restrict__ Dskip,
                                              _Float16* __restrict__ yp16) {
  int b = blockIdx.x;
  int d = threadIdx.x;
  __shared__ float bc[2][2 * D_STATE];
  float Am[D_STATE];
#pragma unroll
  for (int s = 0; s < D_STATE; ++s) Am[s] = -__expf(A_log[d * D_STATE + s]);
  float Dv = Dskip[d];
  float hs[D_STATE];
#pragma unroll
  for (int s = 0; s < D_STATE; ++s) hs[s] = 0.f;

  size_t tok0 = (size_t)b * Lseq;
  if (d < 2 * D_STATE) bc[0][d] = xdbc[tok0 * XDBC_LD + DT_RANK + d];
  float dtc = dt[tok0 * D_INNER + d];
  float uc  = u32[tok0 * D_INNER + d];
  float zc  = xz[tok0 * 512 + D_INNER + d];
  __syncthreads();

  for (int l = 0; l < Lseq; ++l) {
    size_t tok = tok0 + l;
    int cur = l & 1;
    float dtn = 0.f, un = 0.f, zn = 0.f;
    if (l + 1 < Lseq) {                                    // uniform branch
      if (d < 2 * D_STATE) bc[cur ^ 1][d] = xdbc[(tok + 1) * XDBC_LD + DT_RANK + d];
      dtn = dt[(tok + 1) * D_INNER + d];
      un  = u32[(tok + 1) * D_INNER + d];
      zn  = xz[(tok + 1) * 512 + D_INNER + d];
    }
    float du = dtc * uc;
    float y = 0.f;
#pragma unroll
    for (int s = 0; s < D_STATE; ++s) {
      float dA = __expf(dtc * Am[s]);
      hs[s] = hs[s] * dA + du * bc[cur][s];
      y += hs[s] * bc[cur][D_STATE + s];
    }
    float yp = (y + uc * Dv) * fsilu(zc);
    yp16[tok * D_INNER + d] = (_Float16)yp;
    __syncthreads();                                       // bc[cur^1] ready, bc[cur] reusable
    dtc = dtn; uc = un; zc = zn;
  }
}

// ---------------- LSTM scan: per step  g = xg_t + h @ w_hh^T  via WMMA ----------------
// 16 waves: wave = mt(2) x ut(8). Wave handles m-tile mt (16 batches) and unit columns
// ut*16..ut*16+15 across ALL FOUR gates (n-tiles ut, ut+8, ut+16, ut+24) so the gate
// elementwise is wave-local; cell state c and hidden h live in registers (8 per lane).
// Compiler hoists all 16 w_hh B-fragments (128 VGPRs) out of the time loop.
__global__ __launch_bounds__(512) void k_lstm(const float* __restrict__ xg,
                                              const _Float16* __restrict__ whh16,
                                              float* __restrict__ hlast) {
  __shared__ _Float16 hsh[Bsz * LSTM_H];                   // 8 KB, WMMA-A layout friendly
  int tid = threadIdx.x;
  int lane = tid & 31, wave = tid >> 5;
  int mt = wave >> 3;                                      // 0..1 (batch tile)
  int ut = wave & 7;                                       // 0..7 (unit tile)
  int half = lane >> 4;
  int j = ut * 16 + (lane & 15);                           // hidden unit 0..127
  float c_reg[8], h_reg[8];
#pragma unroll
  for (int r = 0; r < 8; ++r) { c_reg[r] = 0.f; h_reg[r] = 0.f; }
  for (int i = tid; i < Bsz * LSTM_H; i += 512) hsh[i] = (_Float16)0.f;
  __syncthreads();

  for (int t = 0; t < Lseq; ++t) {
    v8f acc0 = zero8(), acc1 = zero8(), acc2 = zero8(), acc3 = zero8();
    for (int k0 = 0; k0 < LSTM_H; k0 += 32) {
      v16h a = load_a(&hsh[mt * 16 * LSTM_H + k0], LSTM_H);
      v16h b0 = load_b(whh16 + (size_t)(0 * LSTM_H + ut * 16) * LSTM_H + k0, LSTM_H);
      acc0 = WMMA_F16(a, b0, acc0);
      v16h b1 = load_b(whh16 + (size_t)(1 * LSTM_H + ut * 16) * LSTM_H + k0, LSTM_H);
      acc1 = WMMA_F16(a, b1, acc1);
      v16h b2 = load_b(whh16 + (size_t)(2 * LSTM_H + ut * 16) * LSTM_H + k0, LSTM_H);
      acc2 = WMMA_F16(a, b2, acc2);
      v16h b3 = load_b(whh16 + (size_t)(3 * LSTM_H + ut * 16) * LSTM_H + k0, LSTM_H);
      acc3 = WMMA_F16(a, b3, acc3);
    }
    __syncthreads();  // all waves done reading hsh
#pragma unroll
    for (int r = 0; r < 8; ++r) {
      int b = mt * 16 + half * 8 + r;
      const float* xr = xg + (size_t)(b * Lseq + t) * 512;
      float gi = acc0[r] + xr[j];
      float gf = acc1[r] + xr[LSTM_H + j];
      float gg = acc2[r] + xr[2 * LSTM_H + j];
      float go = acc3[r] + xr[3 * LSTM_H + j];
      float i_ = fsigm(gi), f_ = fsigm(gf), g_ = ftanh(gg), o_ = fsigm(go);
      float c = f_ * c_reg[r] + i_ * g_;
      c_reg[r] = c;
      float h = o_ * ftanh(c);
      h_reg[r] = h;
      hsh[b * LSTM_H + j] = (_Float16)h;
    }
    __syncthreads();  // hsh updated for next step
  }
#pragma unroll
  for (int r = 0; r < 8; ++r) {
    int b = mt * 16 + half * 8 + r;
    hlast[b * LSTM_H + j] = h_reg[r];
  }
}

// ---------------- FC head ----------------
__global__ void k_fc1(const float* __restrict__ h, const float* __restrict__ w,
                      const float* __restrict__ b, float* __restrict__ o) {
  int i = blockIdx.x * blockDim.x + threadIdx.x;           // 32*128
  if (i >= Bsz * LSTM_H) return;
  int bi = i >> 7, jn = i & 127;
  float acc = b[jn];
  const float* hr = h + (size_t)bi * LSTM_H;
  const float* wr = w + (size_t)jn * LSTM_H;
#pragma unroll 8
  for (int k = 0; k < LSTM_H; ++k) acc += hr[k] * wr[k];
  o[i] = fmaxf(acc, 0.f);
}

__global__ void k_fc2(const float* __restrict__ f1, const float* __restrict__ w,
                      const float* __restrict__ b, float* __restrict__ o) {
  int i = blockIdx.x * blockDim.x + threadIdx.x;           // 32*5
  if (i >= Bsz * NUM_CLASSES) return;
  int bi = i / NUM_CLASSES, kn = i % NUM_CLASSES;
  float acc = b[kn];
  const float* fr = f1 + (size_t)bi * 128;
  const float* wr = w + (size_t)kn * 128;
#pragma unroll 8
  for (int k = 0; k < 128; ++k) acc += fr[k] * wr[k];
  o[i] = acc;
}

// ---------------- host launcher ----------------
extern "C" void kernel_launch(void* const* d_in, const int* in_sizes, int n_in,
                              void* d_out, int out_size, void* d_ws, size_t ws_size,
                              hipStream_t stream) {
  (void)in_sizes; (void)n_in; (void)out_size; (void)ws_size;
  const float* x        = (const float*)d_in[0];
  const float* w_proj   = (const float*)d_in[1];
  const float* b_proj   = (const float*)d_in[2];
  const float* w_inproj = (const float*)d_in[3];
  const float* conv_w   = (const float*)d_in[4];
  const float* conv_b   = (const float*)d_in[5];
  const float* w_xproj  = (const float*)d_in[6];
  const float* w_dt     = (const float*)d_in[7];
  const float* b_dt     = (const float*)d_in[8];
  const float* A_log    = (const float*)d_in[9];
  const float* Dskip    = (const float*)d_in[10];
  const float* w_out    = (const float*)d_in[11];
  const float* w_ih     = (const float*)d_in[12];
  const float* w_hh     = (const float*)d_in[13];
  const float* b_ih     = (const float*)d_in[14];
  const float* b_hh     = (const float*)d_in[15];
  const float* w_fc1    = (const float*)d_in[16];
  const float* b_fc1    = (const float*)d_in[17];
  const float* w_fc2    = (const float*)d_in[18];
  const float* b_fc2    = (const float*)d_in[19];
  float* out = (float*)d_out;

  char* ws = (char*)d_ws;
  size_t off = 0;
  auto alloc = [&](size_t bytes) -> char* {
    char* p = ws + off;
    off += (bytes + 255) & ~(size_t)255;
    return p;
  };
  _Float16* h16    = (_Float16*)alloc((size_t)NTOK * 128 * 2);
  float*    xz     = (float*)   alloc((size_t)NTOK * 512 * 4);
  float*    u32b   = (float*)   alloc((size_t)NTOK * 256 * 4);
  _Float16* u16b   = (_Float16*)alloc((size_t)NTOK * 256 * 2);
  float*    xdbc   = (float*)   alloc((size_t)NTOK * XDBC_LD * 4);
  float*    dtb    = (float*)   alloc((size_t)NTOK * 256 * 4);
  _Float16* yp16   = (_Float16*)alloc((size_t)NTOK * 256 * 2);
  _Float16* ymid16 = (_Float16*)alloc((size_t)NTOK * 128 * 2);
  float*    xgb    = (float*)   alloc((size_t)NTOK * 512 * 4);
  float*    hlast  = (float*)   alloc((size_t)Bsz * 128 * 4);
  float*    fc1b   = (float*)   alloc((size_t)Bsz * 128 * 4);
  float*    bsum   = (float*)   alloc(512 * 4);
  _Float16* wip16  = (_Float16*)alloc(512 * 128 * 2);
  _Float16* wxp16  = (_Float16*)alloc(XDBC_LD * 256 * 2);
  _Float16* wout16 = (_Float16*)alloc(128 * 256 * 2);
  _Float16* wih16  = (_Float16*)alloc(512 * 128 * 2);
  _Float16* whh16  = (_Float16*)alloc(512 * 128 * 2);

  // weight prep (f32 -> f16 / padded)
  k_f32_to_f16<<<(512 * 128 + 255) / 256, 256, 0, stream>>>(w_inproj, wip16, 512 * 128);
  k_pad_xproj <<<(XDBC_LD * 256 + 255) / 256, 256, 0, stream>>>(w_xproj, wxp16);
  k_f32_to_f16<<<(128 * 256 + 255) / 256, 256, 0, stream>>>(w_out, wout16, 128 * 256);
  k_f32_to_f16<<<(512 * 128 + 255) / 256, 256, 0, stream>>>(w_ih, wih16, 512 * 128);
  k_f32_to_f16<<<(512 * 128 + 255) / 256, 256, 0, stream>>>(w_hh, whh16, 512 * 128);
  k_bias_sum  <<<2, 256, 0, stream>>>(b_ih, b_hh, bsum, 512);

  // 1) input projection (K=12, scalar)
  k_proj<<<(NTOK * 128) / 256, 256, 0, stream>>>(x, w_proj, b_proj, h16);
  // 2) xz = h @ w_inproj^T   (32768x512, K=128)
  k_gemm<4><<<dim3(NTOK / 64, 512 / 64), 128, 0, stream>>>(h16, 128, wip16, 128, nullptr,
                                                           xz, nullptr, 512, 128);
  // 3) causal conv + SiLU
  k_conv_silu<<<(NTOK * 256) / 256, 256, 0, stream>>>(xz, conv_w, conv_b, u32b, u16b);
  // 4) xdbc = u @ w_xproj^T  (32768x48 padded, K=256)
  k_gemm<3><<<dim3(NTOK / 64, 1), 128, 0, stream>>>(u16b, 256, wxp16, 256, nullptr,
                                                    xdbc, nullptr, XDBC_LD, 256);
  // 5) dt = softplus(...)    (K=8, scalar)
  k_dt<<<(NTOK * 256) / 256, 256, 0, stream>>>(xdbc, w_dt, b_dt, dtb);
  // 6) selective scan (pipelined)
  k_scan<<<Bsz, 256, 0, stream>>>(xdbc, dtb, u32b, xz, A_log, Dskip, yp16);
  // 7) ymid = ypost @ w_out^T (32768x128, K=256)
  k_gemm<4><<<dim3(NTOK / 64, 128 / 64), 128, 0, stream>>>(yp16, 256, wout16, 256, nullptr,
                                                           nullptr, ymid16, 128, 256);
  // 8) xg = ymid @ w_ih^T + (b_ih+b_hh) (32768x512, K=128)
  k_gemm<4><<<dim3(NTOK / 64, 512 / 64), 128, 0, stream>>>(ymid16, 128, wih16, 128, bsum,
                                                           xgb, nullptr, 512, 128);
  // 9) LSTM scan (single WG, WMMA per step)
  k_lstm<<<1, 512, 0, stream>>>(xgb, whh16, hlast);
  // 10) FC head
  k_fc1<<<(Bsz * 128 + 255) / 256, 256, 0, stream>>>(hlast, w_fc1, b_fc1, fc1b);
  k_fc2<<<1, 256, 0, stream>>>(fc1b, w_fc2, b_fc2, out);
}
